// ProjectedAdaptiveLogSoftmax_31645319037261
// MI455X (gfx1250) — matrix operand order repaired and checked by reading, not compiled, not run on hardware
//
#include <hip/hip_runtime.h>
#include <hip/hip_bf16.h>

// ---------------------------------------------------------------------------
// Projected adaptive log-softmax NLL for MI455X (gfx1250, wave32, WMMA bf16)
// Weights staged once per block as bf16 in LDS (320KB/WGP), WMMA from LDS B.
// All tile strides are template constants -> immediate addressing.
// ---------------------------------------------------------------------------

typedef __attribute__((ext_vector_type(16))) __bf16       v16bf;
typedef __attribute__((ext_vector_type(8)))  float        v8f;
typedef __attribute__((ext_vector_type(4)))  unsigned int v4u;
typedef __attribute__((ext_vector_type(2)))  unsigned int v2u;

union BV {
  v16bf v;
  unsigned short u[16];
  v4u q[2];
};

__device__ __forceinline__ unsigned short f2bf(float f) {
  unsigned int x = __float_as_uint(f);
  x += 0x7fffu + ((x >> 16) & 1u);      // round-to-nearest-even
  return (unsigned short)(x >> 16);
}

// (m,s) log-sum-exp running-state merge; -1e30 sentinel keeps this NaN-free
__device__ __forceinline__ void ms_merge(float& m, float& s, float mo, float so) {
  float mn = fmaxf(m, mo);
  s = s * __expf(m - mn) + so * __expf(mo - mn);
  m = mn;
}

// ------------------------------ small prep ---------------------------------

// f32 -> bf16, 4 elements per thread (n must be a multiple of 4)
__global__ void cvt_bf16_kernel(const float* __restrict__ in,
                                unsigned short* __restrict__ out, int n4) {
  int i = blockIdx.x * blockDim.x + threadIdx.x;
  if (i >= n4) return;
  float4 f = ((const float4*)in)[i];
  v2u pk;
  pk.x = (unsigned int)f2bf(f.x) | ((unsigned int)f2bf(f.y) << 16);
  pk.y = (unsigned int)f2bf(f.z) | ((unsigned int)f2bf(f.w) << 16);
  ((v2u*)out)[i] = pk;
}

// f32 [1024, NC] row-major -> bf16 [NC, 1024] (transposed, K contiguous)
template <int NC>
__global__ void cvt_bf16_T_kernel(const float* __restrict__ in,
                                  unsigned short* __restrict__ out) {
  int i = blockIdx.x * blockDim.x + threadIdx.x;
  if (i >= 1024 * NC) return;
  const int k = i / NC, c = i % NC;            // NC is a power of two
  out[(size_t)c * 1024 + k] = f2bf(in[i]);
}

__global__ void zero_u16_kernel(unsigned short* __restrict__ p, int n) {
  int i = blockIdx.x * blockDim.x + threadIdx.x;
  if (i < n) p[i] = 0;
}

// Per-row remapped target columns: [head, tail0, tail1, tail2] * 1024
__global__ void tcol_kernel(const int* __restrict__ target, int* __restrict__ tcol) {
  int r = blockIdx.x * blockDim.x + threadIdx.x;
  if (r >= 1024) return;
  int t = target[r];
  int hc = t, c1 = -1, c2 = -1, c3 = -1;
  if (t >= 20000 && t < 40000)        { hc = 20002; c1 = t - 20000;  }
  else if (t >= 40000 && t < 200000)  { hc = 20001; c2 = t - 40000;  }
  else if (t >= 200000)               { hc = 20000; c3 = t - 200000; }
  tcol[r] = hc; tcol[1024 + r] = c1; tcol[2048 + r] = c2; tcol[3072 + r] = c3;
}

// ------------------------- projection: XP = x @ P --------------------------
// One wave per 16x16 output tile, bf16 WMMA over K=1024.
// PbT is P transposed to [NC, 1024] bf16, so B loads are 2x b128 per k-step.

template <int NC, int KP>
__global__ __launch_bounds__(32) void proj_wmma_kernel(
    const unsigned short* __restrict__ xb,    // x in bf16, [1024,1024]
    const unsigned short* __restrict__ PbT,   // [NC, 1024] bf16 (transposed)
    unsigned short* __restrict__ XP)          // [1024, KP] bf16
{
  const int lane = threadIdx.x & 31;
  const int ln = lane & 15, lg = lane >> 4;
  const unsigned short* arow = xb + (size_t)(blockIdx.x * 16 + ln) * 1024;
  const unsigned short* bcol = PbT + (size_t)(blockIdx.y * 16 + ln) * 1024;

  v8f acc = {};
  for (int k0 = 0; k0 < 1024; k0 += 32) {
    BV a, b;
    a.q[0] = *(const v4u*)(arow + k0 + 8 * lg);
    a.q[1] = *(const v4u*)(arow + k0 + 16 + 8 * lg);
    b.q[0] = *(const v4u*)(bcol + k0 + 8 * lg);
    b.q[1] = *(const v4u*)(bcol + k0 + 16 + 8 * lg);
    acc = __builtin_amdgcn_wmma_f32_16x16x32_bf16(
        false, a.v, false, b.v, (short)0, acc, false, false);
  }
  unsigned short* op =
      XP + (size_t)(blockIdx.x * 16 + lg * 8) * KP + blockIdx.y * 16 + ln;
#pragma unroll
  for (int j = 0; j < 8; ++j)
    op[j * KP] = f2bf(acc[j]);               // constant-offset stores
}

// -------- fused GEMM + online softmax partials over a 128-col stripe -------
// Phase 1: stage the block's 128 weight rows as bf16 into LDS (converted once).
// Phase 2: 8 waves x 16 columns each; loop all 64 row-blocks (4 per pass),
// B tiles from LDS (ds_load_b128, conflict-free via +8 ushort padding),
// A tiles from L2-resident bf16 activations. Emit per-(stripe,row) (m,s)
// partials and capture the target-column logit.

extern __shared__ unsigned short ldsB[];     // [128][KP + 8] ushorts

template <int KP, int KW>
__global__ __launch_bounds__(256) void gemm_softmax_kernel(
    const unsigned short* __restrict__ XP,
    const float* __restrict__ W, const float* __restrict__ b, int split,
    const float* __restrict__ Wx, const float* __restrict__ bx,  // head extra rows
    int Vc, const int* __restrict__ tcol,
    float2* __restrict__ partials, float* __restrict__ tlog)
{
  __shared__ float lm[8][16];
  __shared__ float lss[8][16];

  constexpr int LROW = KP + 8;               // ushorts per LDS column row
  const int tid = threadIdx.x;

  // ---- Phase 1: convert this block's weight stripe into LDS (once) ----
  constexpr int NC4 = KW >> 2;               // float4 chunks per column
  for (int idx = tid; idx < 128 * NC4; idx += 256) {
    const int c  = idx / NC4;                // power-of-two -> shift
    const int k4 = (idx - c * NC4) << 2;
    const int vg = blockIdx.x * 128 + c;
    v2u pk;
    if (vg < Vc) {
      const float* wr = (vg < split) ? (W + (size_t)vg * KW + k4)
                                     : (Wx + (size_t)(vg - split) * KW + k4);
      float4 wf = *(const float4*)wr;
      pk.x = (unsigned int)f2bf(wf.x) | ((unsigned int)f2bf(wf.y) << 16);
      pk.y = (unsigned int)f2bf(wf.z) | ((unsigned int)f2bf(wf.w) << 16);
    } else {
      pk.x = 0u; pk.y = 0u;
    }
    *(v2u*)(&ldsB[c * LROW + k4]) = pk;
  }
  if (KW < KP) {                             // zero K padding (tail3: 16 -> 32)
    constexpr int PAD = KP - KW;
    for (int idx = tid; idx < 128 * PAD; idx += 256) {
      const int c = idx / PAD;
      ldsB[c * LROW + KW + (idx - c * PAD)] = 0;
    }
  }
  __syncthreads();

  // ---- Phase 2: WMMA over all row-blocks ----
  const int wave = tid >> 5, lane = tid & 31;
  const int ln = lane & 15, lg = lane >> 4;
  const int vglob = blockIdx.x * 128 + wave * 16 + ln;
  const bool valid = vglob < Vc;
  float bias = 0.0f;
  if (valid) bias = (vglob < split) ? b[vglob] : bx[vglob - split];
  const unsigned short* bcol = ldsB + (wave * 16 + ln) * LROW;

  for (int rb0 = 0; rb0 < 64; rb0 += 4) {
    const unsigned short* abase = XP + ((size_t)rb0 * 16 + ln) * KP;
    v8f acc[4] = {{}, {}, {}, {}};
    for (int k0 = 0; k0 < KP; k0 += 32) {
      BV bm;
      const unsigned short* bp = bcol + k0 + lg * 16;
      bm.q[0] = *(const v4u*)(bp);           // ds_load_b128
      bm.q[1] = *(const v4u*)(bp + 8);       // ds_load_b128
      const unsigned short* ap = abase + k0 + 16 * lg;
#pragma unroll
      for (int i = 0; i < 4; ++i) {          // 4 row-blocks reuse B registers
        BV a;
        a.q[0] = *(const v4u*)(ap + i * 16 * KP);        // immediate offsets
        a.q[1] = *(const v4u*)(ap + i * 16 * KP + 8);
        acc[i] = __builtin_amdgcn_wmma_f32_16x16x32_bf16(
            false, a.v, false, bm.v, (short)0, acc[i], false, false);
      }
    }
#pragma unroll
    for (int i = 0; i < 4; ++i) {
      const int rowbase = (rb0 + i) * 16 + lg * 8;
      float mj[8], sj[8];
#pragma unroll
      for (int j = 0; j < 8; ++j) {
        float val = acc[i][j] + bias;
        const int row = rowbase + j;
        if (valid && tcol[row] == vglob) tlog[row] = val;  // unique writer
        mj[j] = valid ? val : -1e30f;
        sj[j] = valid ? 1.0f : 0.0f;
      }
      // reduce (m,s) across the 16 lanes of this half-wave (16 columns)
#pragma unroll
      for (int mask = 1; mask <= 8; mask <<= 1) {
#pragma unroll
        for (int j = 0; j < 8; ++j) {
          float mo = __shfl_xor(mj[j], mask, 32);
          float so = __shfl_xor(sj[j], mask, 32);
          ms_merge(mj[j], sj[j], mo, so);
        }
      }
      if (ln == 0) {
#pragma unroll
        for (int j = 0; j < 8; ++j) {
          lm[wave][lg * 8 + j]  = mj[j];
          lss[wave][lg * 8 + j] = sj[j];
        }
      }
      __syncthreads();
      if (tid < 16) {   // merge the 8 waves (128 columns) -> stripe partial
        float M = -1e30f, S = 0.0f;
        for (int w = 0; w < 8; ++w) ms_merge(M, S, lm[w][tid], lss[w][tid]);
        partials[(size_t)blockIdx.x * 1024 + (rb0 + i) * 16 + tid] =
            make_float2(M, S);
      }
      __syncthreads();
    }
  }
}

// -------------------- per-row merge of stripe partials ---------------------

__device__ __forceinline__ float cluster_nll(const float2* __restrict__ p,
                                             int nb, int r, float tl) {
  float M = -1e30f, S = 0.0f;
  for (int bn = 0; bn < nb; ++bn) {
    float2 v = p[(size_t)bn * 1024 + r];
    ms_merge(M, S, v.x, v.y);
  }
  return -(tl - M - __logf(S));
}

__global__ void reduce_kernel(const float2* __restrict__ ph,
                              const float2* __restrict__ p1,
                              const float2* __restrict__ p2,
                              const float2* __restrict__ p3,
                              const float* __restrict__ tlog,
                              const int* __restrict__ target,
                              float* __restrict__ out,
                              int nbh, int nb1, int nb2, int nb3)
{
  int r = blockIdx.x * blockDim.x + threadIdx.x;
  if (r >= 1024) return;
  float nll = cluster_nll(ph, nbh, r, tlog[r]);
  int t = target[r];
  if (t >= 20000 && t < 40000)        nll += cluster_nll(p1, nb1, r, tlog[1024 + r]);
  else if (t >= 40000 && t < 200000)  nll += cluster_nll(p2, nb2, r, tlog[2048 + r]);
  else if (t >= 200000)               nll += cluster_nll(p3, nb3, r, tlog[3072 + r]);
  out[r] = nll;
}

// ------------------------------- launcher ----------------------------------

extern "C" void kernel_launch(void* const* d_in, const int* in_sizes, int n_in,
                              void* d_out, int out_size, void* d_ws, size_t ws_size,
                              hipStream_t stream) {
  (void)in_sizes; (void)n_in; (void)out_size; (void)ws_size;
  const float* x      = (const float*)d_in[0];
  const int*   target = (const int*)  d_in[1];
  const float* W0 = (const float*)d_in[2];
  const float* b0 = (const float*)d_in[3];
  const float* W1 = (const float*)d_in[4];
  const float* b1 = (const float*)d_in[5];
  const float* W2 = (const float*)d_in[6];
  const float* b2 = (const float*)d_in[7];
  const float* W3 = (const float*)d_in[8];
  const float* b3 = (const float*)d_in[9];
  const float* CW = (const float*)d_in[10];
  const float* CB = (const float*)d_in[11];
  const float* P0 = (const float*)d_in[12];
  const float* P1 = (const float*)d_in[13];
  const float* P2 = (const float*)d_in[14];
  const float* P3 = (const float*)d_in[15];
  float* out = (float*)d_out;

  char* ws = (char*)d_ws;
  size_t off = 0;
  auto carve = [&](size_t bytes) {
    char* p = ws + off;
    off = (off + bytes + 255) & ~(size_t)255;
    return p;
  };
  unsigned short* xb  = (unsigned short*)carve((size_t)1024 * 1024 * 2);
  unsigned short* xp0 = (unsigned short*)carve((size_t)1024 * 1024 * 2);
  unsigned short* xp1 = (unsigned short*)carve((size_t)1024 * 256 * 2);
  unsigned short* xp2 = (unsigned short*)carve((size_t)1024 * 64 * 2);
  unsigned short* xp3 = (unsigned short*)carve((size_t)1024 * 32 * 2);  // K padded 16->32
  unsigned short* pb0 = (unsigned short*)carve((size_t)1024 * 1024 * 2);  // transposed
  unsigned short* pb1 = (unsigned short*)carve((size_t)256  * 1024 * 2);
  unsigned short* pb2 = (unsigned short*)carve((size_t)64   * 1024 * 2);
  unsigned short* pb3 = (unsigned short*)carve((size_t)16   * 1024 * 2);
  int*   tcol = (int*)  carve((size_t)4 * 1024 * 4);
  float* tlog = (float*)carve((size_t)4 * 1024 * 4);
  const int NBH = (20003  + 127) / 128;   // 157
  const int NB1 = (20000  + 127) / 128;   // 157
  const int NB2 = (160000 + 127) / 128;   // 1250
  const int NB3 = (67735  + 127) / 128;   // 530
  float2* ph = (float2*)carve((size_t)NBH * 1024 * sizeof(float2));
  float2* p1 = (float2*)carve((size_t)NB1 * 1024 * sizeof(float2));
  float2* p2 = (float2*)carve((size_t)NB2 * 1024 * sizeof(float2));
  float2* p3 = (float2*)carve((size_t)NB3 * 1024 * sizeof(float2));

  // Allow >64KB dynamic LDS for the head stripe (258KB; CDNA5: 320KB/WG).
  (void)hipFuncSetAttribute((const void*)gemm_softmax_kernel<1024, 1024>,
                            hipFuncAttributeMaxDynamicSharedMemorySize,
                            128 * (1024 + 8) * 2);
  (void)hipFuncSetAttribute((const void*)gemm_softmax_kernel<256, 256>,
                            hipFuncAttributeMaxDynamicSharedMemorySize,
                            128 * (256 + 8) * 2);

  cvt_bf16_kernel<<<1024, 256, 0, stream>>>(x, xb, 1024 * 1024 / 4);
  cvt_bf16_T_kernel<1024><<<4096, 256, 0, stream>>>(P0, pb0);
  cvt_bf16_T_kernel<256><<<1024, 256, 0, stream>>>(P1, pb1);
  cvt_bf16_T_kernel<64><<<256,  256, 0, stream>>>(P2, pb2);
  cvt_bf16_T_kernel<16><<<64,   256, 0, stream>>>(P3, pb3);
  zero_u16_kernel<<<128, 256, 0, stream>>>(xp3, 1024 * 32);
  tcol_kernel<<<4, 256, 0, stream>>>(target, tcol);

  proj_wmma_kernel<1024, 1024><<<dim3(64, 64), 32, 0, stream>>>(xb, pb0, xp0);
  proj_wmma_kernel<256,  256 ><<<dim3(64, 16), 32, 0, stream>>>(xb, pb1, xp1);
  proj_wmma_kernel<64,   64  ><<<dim3(64,  4), 32, 0, stream>>>(xb, pb2, xp2);
  proj_wmma_kernel<16,   32  ><<<dim3(64,  1), 32, 0, stream>>>(xb, pb3, xp3);

  const size_t shH = (size_t)128 * (1024 + 8) * 2;  // 264192 B
  const size_t sh1 = (size_t)128 * (256  + 8) * 2;  //  67584 B
  const size_t sh2 = (size_t)128 * (64   + 8) * 2;  //  18432 B
  const size_t sh3 = (size_t)128 * (32   + 8) * 2;  //  10240 B
  gemm_softmax_kernel<1024, 1024><<<NBH, 256, shH, stream>>>(
      xp0, W0, b0, 20000, CW, CB, 20003, tcol, ph, tlog);
  gemm_softmax_kernel<256, 256><<<NB1, 256, sh1, stream>>>(
      xp1, W1, b1, 20000, W1, b1, 20000, tcol + 1024, p1, tlog + 1024);
  gemm_softmax_kernel<64, 64><<<NB2, 256, sh2, stream>>>(
      xp2, W2, b2, 160000, W2, b2, 160000, tcol + 2048, p2, tlog + 2048);
  gemm_softmax_kernel<32, 16><<<NB3, 256, sh3, stream>>>(
      xp3, W3, b3, 67735, W3, b3, 67735, tcol + 3072, p3, tlog + 3072);

  reduce_kernel<<<4, 256, 0, stream>>>(ph, p1, p2, p3, tlog, target, out,
                                       NBH, NB1, NB2, NB3);
}
